// HungarianMatcher_51118700757462
// MI455X (gfx1250) — matrix operand order repaired and checked by reading, not compiled, not run on hardware
//
#include <hip/hip_runtime.h>
#include <hip/hip_bf16.h>
#include <math.h>

// Problem constants (match reference)
#define B_ 64
#define Q_ 512
#define C_ 128
#define P_ 2048
#define G_ 256

#define TQ 128        // q rows per block
#define KSTEP 32      // p (contraction) per k-iteration
#define NKITER (P_ / KSTEP)
#define NTHREADS 512  // 16 waves (wave32)

#define A_STRIDE 40   // bf16 elems per As row (128 rows), padded: 80B, 16B-aligned, conflict-free
#define B_STRIDE 264  // bf16 elems per Bs row (32 rows), padded: 528B, 16B-aligned

typedef __attribute__((ext_vector_type(16))) __bf16 bf16x16;
typedef __attribute__((ext_vector_type(8)))  __bf16 bf16x8;
typedef __attribute__((ext_vector_type(8)))  float  f32x8;

__device__ __forceinline__ float softplusf(float x) {
    // numerically stable softplus
    return fmaxf(x, 0.0f) + log1pf(__expf(-fabsf(x)));
}

__global__ __launch_bounds__(NTHREADS)
void hungarian_cost_fused_wmma(const float* __restrict__ logits,   // [B,Q,C]
                               const float* __restrict__ attw,     // [B,P,Q]
                               const float* __restrict__ oneh,     // [B,P,G]
                               const int*   __restrict__ ids,      // [B,G]
                               float* __restrict__ out) {          // [B,Q,G]
    __shared__ __bf16 As[TQ * A_STRIDE];        // A tile, transposed: [q][p]
    __shared__ __bf16 Bs[KSTEP * B_STRIDE];     // B tile, natural:    [p][g]
    __shared__ float  redG[16][TQ];             // per-wave grouping-softplus partials
    __shared__ float  redA[4][TQ];              // activity-softplus partials
    __shared__ float  rowterm[TQ];              // sp_g + sp_a per local q

    const int t    = threadIdx.x;
    const int b    = blockIdx.y;
    const int q0   = blockIdx.x * TQ;
    const int lane = t & 31;
    const int wave = t >> 5;
    const int wm   = wave >> 3;   // 0..1  -> 64 q rows each
    const int wn   = wave & 7;    // 0..7  -> 32 g cols each

    // ---- prologue: activity softplus, deterministic partials -------------
    {
        const int row = t >> 2;   // 0..127 (local q)
        const int qtr = t & 3;    // 0..3
        const float* lp = logits + ((size_t)b * Q_ + (q0 + row)) * C_ + qtr * 32;
        float s = 0.0f;
        #pragma unroll 8
        for (int c = 0; c < 32; ++c) s += softplusf(lp[c]);
        redA[qtr][row] = s;
    }

    // accumulators: 4 (m) x 2 (n) tiles of 16x16 f32
    f32x8 acc[4][2];
    #pragma unroll
    for (int tm = 0; tm < 4; ++tm)
        #pragma unroll
        for (int tn = 0; tn < 2; ++tn) { f32x8 z = {}; acc[tm][tn] = z; }

    // per-thread grouping-softplus partials for 4 fixed local-q columns
    const int qcol = (t & 31) * 4;             // same 4 q columns every iteration
    float sp0 = 0.f, sp1 = 0.f, sp2 = 0.f, sp3 = 0.f;

    // Per-thread source pointers (fixed lane pattern, advance by KSTEP rows/iter)
    const int prowA = t >> 5;                  // A rows prowA, prowA+16
    const int prowB = t >> 6;                  // B rows prowB + 8*i
    const int gcB   = (t & 63) * 4;
    const float* aPtr = attw + ((size_t)b * P_ + prowA) * Q_ + q0 + qcol;
    const float* bPtr = oneh + ((size_t)b * P_ + prowB) * G_ + gcB;

    // ---- software pipeline: prefetch tile 0 into registers ---------------
    float4 aReg[2];
    float4 bReg[4];
    #pragma unroll
    for (int i = 0; i < 2; ++i)
        aReg[i] = *(const float4*)(aPtr + (size_t)(i * 16) * Q_);
    #pragma unroll
    for (int i = 0; i < 4; ++i)
        bReg[i] = *(const float4*)(bPtr + (size_t)(i * 8) * G_);

    // ---- main K loop over P ----------------------------------------------
    for (int ks = 0; ks < NKITER; ++ks) {
        // stage current registers -> LDS (convert to bf16; A transposed),
        // accumulating grouping softplus on the fly
        #pragma unroll
        for (int i = 0; i < 2; ++i) {
            const int prow = prowA + i * 16;
            const float4 v = aReg[i];
            sp0 += softplusf(v.x); sp1 += softplusf(v.y);
            sp2 += softplusf(v.z); sp3 += softplusf(v.w);
            As[(qcol + 0) * A_STRIDE + prow] = (__bf16)v.x;
            As[(qcol + 1) * A_STRIDE + prow] = (__bf16)v.y;
            As[(qcol + 2) * A_STRIDE + prow] = (__bf16)v.z;
            As[(qcol + 3) * A_STRIDE + prow] = (__bf16)v.w;
        }
        #pragma unroll
        for (int i = 0; i < 4; ++i) {
            const int prow = prowB + i * 8;
            const float4 v = bReg[i];
            __bf16* dst = &Bs[prow * B_STRIDE + gcB];
            dst[0] = (__bf16)v.x; dst[1] = (__bf16)v.y;
            dst[2] = (__bf16)v.z; dst[3] = (__bf16)v.w;
        }
        __syncthreads();

        // prefetch next tile into registers: overlaps with ds frag loads + WMMA
        if (ks + 1 < NKITER) {
            const size_t adv = (size_t)(ks + 1) * KSTEP;
            #pragma unroll
            for (int i = 0; i < 2; ++i)
                aReg[i] = *(const float4*)(aPtr + (adv + i * 16) * Q_);
            #pragma unroll
            for (int i = 0; i < 4; ++i)
                bReg[i] = *(const float4*)(bPtr + (adv + i * 8) * G_);
        }

        // A fragments: lane L(0-15): m=L, K=0..7 & 16..23 ; lanes 16-31: K=8..15 & 24..31
        const int m  = lane & 15;
        const int kh = lane >> 4;
        bf16x16 afrag[4];
        #pragma unroll
        for (int tm = 0; tm < 4; ++tm) {
            const __bf16* ap = &As[(wm * 64 + tm * 16 + m) * A_STRIDE];
            bf16x8 lo = *(const bf16x8*)(ap + kh * 8);
            bf16x8 hi = *(const bf16x8*)(ap + 16 + kh * 8);
            afrag[tm] = __builtin_shufflevector(lo, hi,
                0,1,2,3,4,5,6,7,8,9,10,11,12,13,14,15);
        }
        // B fragments: lane L(0-15): n=L, K=0..7 & 16..23 ; lanes 16-31: K=8..15 & 24..31
        bf16x16 bfrag[2];
        #pragma unroll
        for (int tn = 0; tn < 2; ++tn) {
            const __bf16* bp = &Bs[(kh * 8) * B_STRIDE + wn * 32 + tn * 16 + m];
            #pragma unroll
            for (int s = 0; s < 8; ++s) {
                bfrag[tn][s]     = bp[s * B_STRIDE];
                bfrag[tn][s + 8] = bp[(16 + s) * B_STRIDE];
            }
        }

        #pragma unroll
        for (int tm = 0; tm < 4; ++tm)
            #pragma unroll
            for (int tn = 0; tn < 2; ++tn)
                acc[tm][tn] = __builtin_amdgcn_wmma_f32_16x16x32_bf16(
                    /*neg_a=*/false, afrag[tm],
                    /*neg_b=*/false, bfrag[tn],
                    /*c_mod=*/(short)0, acc[tm][tn],
                    /*reuse_a=*/false, /*reuse_b=*/false);
        __syncthreads();
    }

    // ---- combine row terms deterministically -----------------------------
    redG[t >> 5][qcol + 0] = sp0;
    redG[t >> 5][qcol + 1] = sp1;
    redG[t >> 5][qcol + 2] = sp2;
    redG[t >> 5][qcol + 3] = sp3;
    __syncthreads();
    if (t < TQ) {
        float sg = 0.0f;
        #pragma unroll
        for (int w = 0; w < 16; ++w) sg += redG[w][t];
        float sa = redA[0][t] + redA[1][t] + redA[2][t] + redA[3][t];
        rowterm[t] = sg * (1.0f / P_) + sa * (1.0f / C_);
    }
    __syncthreads();

    // ---- epilogue: cost = rowterm - acc/P - logits[b,q,ids[b,g]]/C -------
    const int   n     = lane & 15;
    const int   mhalf = (lane >> 4) * 8;
    const int   g0    = wn * 32;
    const int   ida   = ids[b * G_ + g0 + n];        // tn = 0 column class id
    const int   idb   = ids[b * G_ + g0 + 16 + n];   // tn = 1 column class id
    const float invP  = 1.0f / P_;
    const float invC  = 1.0f / C_;

    #pragma unroll
    for (int tm = 0; tm < 4; ++tm) {
        #pragma unroll
        for (int r = 0; r < 8; ++r) {
            const int    ql    = wm * 64 + tm * 16 + mhalf + r;   // local q
            const size_t qglob = (size_t)b * Q_ + q0 + ql;
            const float  rt    = rowterm[ql];
            const float  ga    = logits[qglob * C_ + ida];
            const float  gb    = logits[qglob * C_ + idb];
            out[qglob * G_ + g0 + n]      = rt - acc[tm][0][r] * invP - ga * invC;
            out[qglob * G_ + g0 + 16 + n] = rt - acc[tm][1][r] * invP - gb * invC;
        }
    }
}

extern "C" void kernel_launch(void* const* d_in, const int* in_sizes, int n_in,
                              void* d_out, int out_size, void* d_ws, size_t ws_size,
                              hipStream_t stream) {
    (void)in_sizes; (void)n_in; (void)d_ws; (void)ws_size; (void)out_size;
    const float* logits = (const float*)d_in[0];   // [B,Q,C]
    const float* attw   = (const float*)d_in[1];   // [B,P,Q]
    const float* oneh   = (const float*)d_in[2];   // [B,P,G]
    const int*   ids    = (const int*)d_in[3];     // [B,G]
    float* out = (float*)d_out;                    // [B,Q,G]

    dim3 grid(Q_ / TQ, B_);   // 4 q-tiles x 64 batches = 256 workgroups
    hungarian_cost_fused_wmma<<<grid, NTHREADS, 0, stream>>>(logits, attw, oneh, ids, out);
}
